// ACFModel_65712999629189
// MI455X (gfx1250) — matrix operand order repaired and checked by reading, not compiled, not run on hardware
//
#include <hip/hip_runtime.h>

// Problem constants (match reference)
#define NPOS 512
#define NREG 49
#define FEATC 2048
#define DW 64
#define FDIM 200

typedef __attribute__((ext_vector_type(16))) __bf16 v16bf;
typedef __attribute__((ext_vector_type(8)))  float  v8f;
typedef __attribute__((ext_vector_type(16))) unsigned short v16us;
typedef __attribute__((ext_vector_type(8)))  unsigned short v8us;

__device__ __forceinline__ unsigned short f2bf(float f) {
  unsigned int u = __float_as_uint(f);
  u += 0x7FFFu + ((u >> 16) & 1u);          // round-to-nearest-even
  return (unsigned short)(u >> 16);
}
__device__ __forceinline__ float bf2f(unsigned short h) {
  return __uint_as_float(((unsigned int)h) << 16);
}

// ---------------- workspace layout (bytes) ----------------
#define WS_CVEC  0u                          // 64 f32: gu@Wc0_u + bc0
#define WS_IVEC  256u                        // 64 f32: gu@Wi0_u + bi0
#define WS_BPACK 512u                        // 2048*64 bf16 fragment-packed Wc0_i
#define WS_ALLX  (512u + 262144u)            // 512*2048 f32
#define WS_LOGIT (WS_ALLX + (unsigned)NPOS*FEATC*4u)  // 512 f32
// total ~4.46 MB

// ---------------- kernel 0: user-dependent bias vectors ----------------
__global__ void acf_prep_vecs(const float* __restrict__ Gu,
                              const float* __restrict__ Wc0_u,
                              const float* __restrict__ bc0,
                              const float* __restrict__ Wi0_u,
                              const float* __restrict__ bi0,
                              const int* __restrict__ user,
                              float* __restrict__ ws_cvec,
                              float* __restrict__ ws_ivec) {
  int d = threadIdx.x;                       // 0..63
  const float* gu = Gu + (long)user[0] * FDIM;
  float c = bc0[d], iv = bi0[d];
  for (int f = 0; f < FDIM; ++f) {
    float g = gu[f];
    c  += g * Wc0_u[f * DW + d];
    iv += g * Wi0_u[f * DW + d];
  }
  ws_cvec[d] = c;
  ws_ivec[d] = iv;
}

// ---------------- kernel 0b: pack Wc0_i (fp32 -> bf16, WMMA-B fragment order) ----------------
// B 32x16 bf16 layout: lanes 0-15 hold K=0..7,16..23, lanes 16-31 hold K=8..15,24..31 (elem e<8 -> K=8h+e, e>=8 -> K=16+8h+e-8).
// Linear index: (((kstep*4 + ntile)*32 + lane)*16 + e), so each lane's fragment is 32 contiguous bytes.
__global__ void acf_pack_B(const float* __restrict__ Wc0_i,
                           unsigned short* __restrict__ wsB) {
  int idx = blockIdx.x * blockDim.x + threadIdx.x;   // over FEATC*DW
  int K = idx >> 6, N = idx & 63;
  int kk = K >> 5, kl = K & 31;
  int lane = (((kl >> 3) & 1) << 4) | (N & 15);
  int e = (kl & 7) | ((kl >> 4) << 3), nt = N >> 4;
  wsB[(((kk * 4 + nt) * 32 + lane) << 4) + e] = f2bf(Wc0_i[idx]);
}

// ---------------- kernel 1: fused component attention (the big one) ----------------
// One block per positive item p. 8 waves = 4 Mtiles x 4 Ntiles of 16x16, K=2048 in 64 steps.
#define SM_F     0u
#define SM_H     (64u * FEATC * 2u)          // 262144 (64 padded rows of bf16 f tile)
#define SM_CV    (SM_H + DW * DW * 4u)       // 278528
#define SM_LG    (SM_CV + 256u)              // 278784
#define SM_WSM   (SM_LG + 256u)              // 279040
#define SMEM1_BYTES (SM_WSM + 256u)          // 279296 (< 320 KB/WGP)

__global__ __launch_bounds__(256)
void acf_component_attn(const float* __restrict__ f_u_i_pos,
                        const unsigned short* __restrict__ wsB,
                        const float* __restrict__ ws_cvec,
                        const float* __restrict__ Wc1,
                        const float* __restrict__ bc1,
                        float* __restrict__ ws_allx) {
  extern __shared__ unsigned char smem[];
  unsigned short* lds_f = (unsigned short*)(smem + SM_F);
  float* hbuf = (float*)(smem + SM_H);
  float* cv   = (float*)(smem + SM_CV);
  float* lgt  = (float*)(smem + SM_LG);
  float* wsm  = (float*)(smem + SM_WSM);

  const int p    = blockIdx.x;
  const int tid  = threadIdx.x;
  const int lane = tid & 31;
  const int w    = tid >> 5;

  // ---- phase 0: stage f[p] (49x2048 fp32) -> LDS bf16, one pass over HBM ----
  const float4* src = (const float4*)(f_u_i_pos + (size_t)p * NREG * FEATC);
  for (int i = tid; i < NREG * FEATC / 4; i += 256) {
    float4 v = src[i];
    unsigned long long pk =
        (unsigned long long)f2bf(v.x)
      | ((unsigned long long)f2bf(v.y) << 16)
      | ((unsigned long long)f2bf(v.z) << 32)
      | ((unsigned long long)f2bf(v.w) << 48);
    *(unsigned long long*)(&lds_f[i * 4]) = pk;
  }
  if (tid < DW) cv[tid] = ws_cvec[tid];
  __syncthreads();

  // ---- phase 1: [64pad x 2048] @ [2048 x 64] via v_wmma_f32_16x16x32_bf16 ----
  const int n0   = (w & 3) << 4;             // N tile
  const int m0   = (w >> 2) << 5;            // two M tiles: m0, m0+16
  const int khiA = ((lane >> 4) & 1) << 3;   // A: lane half selects K +0 / +8 (and +16/+24)
  const int r0   = m0 + (lane & 15);
  const int r1   = r0 + 16;
  v8f c0 = {}, c1 = {};
  for (int kk = 0; kk < FEATC / 32; ++kk) {
    // B fragment: one contiguous 32-byte load per lane from packed ws
    v16us braw = *((const v16us*)wsB + ((kk * 4 + (n0 >> 4)) * 32 + lane));
    v16bf bfr = __builtin_bit_cast(v16bf, braw);

    // A fragments: two 16-byte ds loads each (K groups +0..7 and +16..23 of lane-half base)
    const unsigned short* a0p = lds_f + r0 * FEATC + kk * 32 + khiA;
    const unsigned short* a1p = lds_f + r1 * FEATC + kk * 32 + khiA;
    v8us a0lo = *(const v8us*)a0p;
    v8us a0hi = *(const v8us*)(a0p + 16);
    v8us a1lo = *(const v8us*)a1p;
    v8us a1hi = *(const v8us*)(a1p + 16);
    v16us a0raw, a1raw;
#pragma unroll
    for (int i = 0; i < 8; ++i) {
      a0raw[i] = a0lo[i]; a0raw[i + 8] = a0hi[i];
      a1raw[i] = a1lo[i]; a1raw[i + 8] = a1hi[i];
    }
    v16bf a0 = __builtin_bit_cast(v16bf, a0raw);
    v16bf a1 = __builtin_bit_cast(v16bf, a1raw);
    c0 = __builtin_amdgcn_wmma_f32_16x16x32_bf16(false, a0, false, bfr, (short)0, c0, false, false);
    c1 = __builtin_amdgcn_wmma_f32_16x16x32_bf16(false, a1, false, bfr, (short)0, c1, false, false);
  }

  // ---- epilogue: C layout (VGPR j: M = j + 8*(lane>=16), N = lane&15) + bias + relu ----
  {
    const int col  = n0 + (lane & 15);
    const int rofs = ((lane >> 4) & 1) << 3;
    const float cc = cv[col];
#pragma unroll
    for (int j = 0; j < 8; ++j) {
      float v0 = c0[j] + cc; v0 = v0 > 0.f ? v0 : 0.f;
      hbuf[(m0 + j + rofs) * DW + col] = v0;
      float v1 = c1[j] + cc; v1 = v1 > 0.f ? v1 : 0.f;
      hbuf[(m0 + 16 + j + rofs) * DW + col] = v1;
    }
  }
  __syncthreads();

  // ---- phase 2: region logits + softmax over R=49 ----
  if (tid < NREG) {
    float lg = bc1[0];
    const float* hr = hbuf + tid * DW;
#pragma unroll
    for (int d = 0; d < DW; ++d) lg += hr[d] * Wc1[d];
    lgt[tid] = lg;
  }
  __syncthreads();
  if (tid == 0) {
    float mx = lgt[0];
    for (int r = 1; r < NREG; ++r) mx = lgt[r] > mx ? lgt[r] : mx;
    float s = 0.f;
    for (int r = 0; r < NREG; ++r) { float e = __expf(lgt[r] - mx); wsm[r] = e; s += e; }
    float inv = 1.f / s;
    for (int r = 0; r < NREG; ++r) wsm[r] *= inv;
  }
  __syncthreads();

  // ---- phase 3: all_x[p,:] = sum_r w[r] * f[p,r,:] (f reused from LDS) ----
  float* outp = ws_allx + (size_t)p * FEATC;
  for (int k = tid; k < FEATC; k += 256) {
    float acc = 0.f;
    for (int r = 0; r < NREG; ++r) acc += wsm[r] * bf2f(lds_f[r * FEATC + k]);
    outp[k] = acc;
  }
}

// ---------------- kernel 2: item-level pre-activations + logits ----------------
__global__ __launch_bounds__(64)
void acf_item_logits(const float* __restrict__ Gi, const float* __restrict__ Pi,
                     const float* __restrict__ Wi0_iv, const float* __restrict__ Wi0_ip,
                     const float* __restrict__ Wi0_ix,
                     const float* __restrict__ Wi1, const float* __restrict__ bi1,
                     const float* __restrict__ ws_ivec, const float* __restrict__ ws_allx,
                     const int* __restrict__ user_pos, float* __restrict__ ws_logit) {
  __shared__ float red[DW];
  const int p = blockIdx.x, d = threadIdx.x;
  const long idx = user_pos[p];
  const float* gp = Gi + idx * FDIM;
  const float* pp = Pi + idx * FDIM;
  float acc = ws_ivec[d];
  for (int f = 0; f < FDIM; ++f)
    acc += gp[f] * Wi0_iv[f * DW + d] + pp[f] * Wi0_ip[f * DW + d];
  const float* ax = ws_allx + (size_t)p * FEATC;
  for (int k = 0; k < FEATC; ++k)
    acc += ax[k] * Wi0_ix[k * DW + d];
  acc = acc > 0.f ? acc : 0.f;
  red[d] = acc * Wi1[d];
  __syncthreads();
  if (d == 0) {
    float s = bi1[0];
    for (int i = 0; i < DW; ++i) s += red[i];
    ws_logit[p] = s;
  }
}

// ---------------- kernel 3: softmax over P, all_a, xui, outputs ----------------
__global__ __launch_bounds__(256)
void acf_finalize(const float* __restrict__ Gu, const float* __restrict__ Gi,
                  const float* __restrict__ Pi,
                  const int* __restrict__ user, const int* __restrict__ item,
                  const int* __restrict__ user_pos,
                  const float* __restrict__ ws_logit, float* __restrict__ out) {
  __shared__ float a[NPOS];
  __shared__ float red[256];
  __shared__ float stats[2];
  const int tid = threadIdx.x;
  for (int p = tid; p < NPOS; p += 256) a[p] = ws_logit[p];
  __syncthreads();
  if (tid == 0) {
    float mx = a[0];
    for (int p = 1; p < NPOS; ++p) mx = a[p] > mx ? a[p] : mx;
    float s = 0.f;
    for (int p = 0; p < NPOS; ++p) s += __expf(a[p] - mx);
    stats[0] = mx; stats[1] = 1.f / s;
  }
  __syncthreads();
  const float mx = stats[0], inv = stats[1];
  for (int p = tid; p < NPOS; p += 256) a[p] = __expf(a[p] - mx) * inv;
  __syncthreads();
  const int u = user[0], it = item[0];
  float part = 0.f;
  if (tid < FDIM) {
    float alla = 0.f;
    for (int p = 0; p < NPOS; ++p)
      alla += a[p] * Pi[(long)user_pos[p] * FDIM + tid];
    float gu  = Gu[(long)u * FDIM + tid];
    float gi  = Gi[(long)it * FDIM + tid];
    float pii = Pi[(long)it * FDIM + tid];
    out[1 + tid] = gu;                 // g_u
    out[1 + FDIM + tid] = gi;          // gamma_i
    out[1 + 2 * FDIM + tid] = pii;     // p_i_item
    part = (gu + alla) * gi;
  }
  red[tid] = part;
  __syncthreads();
  if (tid == 0) {
    float s = 0.f;
    for (int i = 0; i < 256; ++i) s += red[i];
    out[0] = s;                        // xui
  }
}

// ---------------- host ----------------
extern "C" void kernel_launch(void* const* d_in, const int* in_sizes, int n_in,
                              void* d_out, int out_size, void* d_ws, size_t ws_size,
                              hipStream_t stream) {
  const float* f      = (const float*)d_in[0];
  const float* Gu     = (const float*)d_in[1];
  const float* Gi     = (const float*)d_in[2];
  const float* Pi     = (const float*)d_in[3];
  const float* Wc0_u  = (const float*)d_in[4];
  const float* Wc0_i  = (const float*)d_in[5];
  const float* bc0    = (const float*)d_in[6];
  const float* Wc1    = (const float*)d_in[7];
  const float* bc1    = (const float*)d_in[8];
  const float* Wi0_u  = (const float*)d_in[9];
  const float* Wi0_iv = (const float*)d_in[10];
  const float* Wi0_ip = (const float*)d_in[11];
  const float* Wi0_ix = (const float*)d_in[12];
  const float* bi0    = (const float*)d_in[13];
  const float* Wi1    = (const float*)d_in[14];
  const float* bi1    = (const float*)d_in[15];
  const int* user     = (const int*)d_in[16];
  const int* item     = (const int*)d_in[17];
  const int* user_pos = (const int*)d_in[18];
  float* out = (float*)d_out;

  unsigned char* ws = (unsigned char*)d_ws;
  float* ws_cvec = (float*)(ws + WS_CVEC);
  float* ws_ivec = (float*)(ws + WS_IVEC);
  unsigned short* wsB = (unsigned short*)(ws + WS_BPACK);
  float* ws_allx = (float*)(ws + WS_ALLX);
  float* ws_logit = (float*)(ws + WS_LOGIT);

  (void)in_sizes; (void)n_in; (void)out_size; (void)ws_size;

  // allow > 64 KB dynamic LDS (CDNA5: 320 KB per WGP)
  hipFuncSetAttribute(reinterpret_cast<const void*>(acf_component_attn),
                      hipFuncAttributeMaxDynamicSharedMemorySize, (int)SMEM1_BYTES);

  acf_prep_vecs<<<1, 64, 0, stream>>>(Gu, Wc0_u, bc0, Wi0_u, bi0, user, ws_cvec, ws_ivec);
  acf_pack_B<<<(FEATC * DW) / 256, 256, 0, stream>>>(Wc0_i, wsB);
  acf_component_attn<<<NPOS, 256, SMEM1_BYTES, stream>>>(f, wsB, ws_cvec, Wc1, bc1, ws_allx);
  acf_item_logits<<<NPOS, 64, 0, stream>>>(Gi, Pi, Wi0_iv, Wi0_ip, Wi0_ix, Wi1, bi1,
                                           ws_ivec, ws_allx, user_pos, ws_logit);
  acf_finalize<<<1, 256, 0, stream>>>(Gu, Gi, Pi, user, item, user_pos, ws_logit, out);
}
